// ForceField_19731079758688
// MI455X (gfx1250) — matrix operand order repaired and checked by reading, not compiled, not run on hardware
//
#include <hip/hip_runtime.h>

typedef __attribute__((ext_vector_type(16))) _Float16 v16h;
typedef __attribute__((ext_vector_type(8)))  float    v8f;

#define L_DIM 64
#define R_DIM 4096
#define E_DIM 32
#define F_DIM 64
#define P_DIM 8
#define EPS_C 1e-10f

__device__ inline v16h cvt16(float4 u0, float4 u1, float4 u2, float4 u3) {
    union { v16h v; _Float16 h[16]; } t;
    t.h[0]  = (_Float16)u0.x; t.h[1]  = (_Float16)u0.y;
    t.h[2]  = (_Float16)u0.z; t.h[3]  = (_Float16)u0.w;
    t.h[4]  = (_Float16)u1.x; t.h[5]  = (_Float16)u1.y;
    t.h[6]  = (_Float16)u1.z; t.h[7]  = (_Float16)u1.w;
    t.h[8]  = (_Float16)u2.x; t.h[9]  = (_Float16)u2.y;
    t.h[10] = (_Float16)u2.z; t.h[11] = (_Float16)u2.w;
    t.h[12] = (_Float16)u3.x; t.h[13] = (_Float16)u3.y;
    t.h[14] = (_Float16)u3.z; t.h[15] = (_Float16)u3.w;
    return t.v;
}

// One wave per 16x16 (l,r) tile. 8 waves/block, 128 blocks -> 1024 tiles.
__global__ __launch_bounds__(256) void ff_main(const float* __restrict__ lig_feat,
                                               const float* __restrict__ rec_feat,
                                               const float* __restrict__ lig_coords,
                                               const float* __restrict__ rec_coord,
                                               float* __restrict__ partial) {
    const int lane = threadIdx.x & 31;
    const int wave = threadIdx.x >> 5;
    const int task = blockIdx.x * 8 + wave;          // 0..1023
    const int l_tile = task & 3;                     // L/16 = 4
    const int r_tile = task >> 2;                    // R/16 = 256
    const int l_base = l_tile * 16;
    const int r_base = r_tile * 16;
    const int half   = lane >> 4;                    // 0 or 1
    const int lm     = lane & 15;

    // exp(-z^2) = 2^(-(z*sqrt(log2 e))^2): fold log2(e) into the distance scale
    // so the HW base-2 exp needs no extra multiply per element.
    const float SQRT_LOG2E = 1.2011224087864498f;    // sqrt(1/ln 2)
    const float cfac   = (32.0f / 10.0f) * SQRT_LOG2E;   // |1/sigma| * sqrt(log2 e)
    const float mu_stc = (10.0f / 31.0f) * cfac;         // linspace step, pre-scaled

    // ---- scaled distances for this lane's 8 (M rows) x 8 (P poses), reused over all e ----
    const int r = r_base + lm;                       // C-layout: N = lane&15
    const float rcx = rec_coord[3 * r + 0];
    const float rcy = rec_coord[3 * r + 1];
    const float rcz = rec_coord[3 * r + 2];

    float dsc[8][8];
#pragma unroll
    for (int i = 0; i < 8; ++i) {
        const int l = l_base + i + 8 * half;         // C-layout: M = i + 8*(lane>=16)
#pragma unroll
        for (int p = 0; p < 8; ++p) {
            const float* lc = lig_coords + ((size_t)p * L_DIM + l) * 3;
            const float dx = lc[0] - rcx;
            const float dy = lc[1] - rcy;
            const float dz = lc[2] - rcz;
            dsc[i][p] = sqrtf((dx * dx + EPS_C) + (dy * dy + EPS_C) + (dz * dz + EPS_C)) * cfac;
        }
    }

    float acc[8];
#pragma unroll
    for (int p = 0; p < 8; ++p) acc[p] = 0.0f;

    // A fragment (16x32 f16): lane = M (both halves), K chunks kb..kb+7 and kb+16..kb+23
    const int akb = half * 8;
    // B fragment (32x16 f16): lane = N, K chunks kb..kb+7 and kb+16..kb+23
    const int bkb = half * 8;
    const float* Abase = lig_feat + (size_t)(l_base + lm) * E_DIM * F_DIM + akb;
    const float* Bbase = rec_feat + (size_t)(r_base + lm) * E_DIM * F_DIM + bkb;

#pragma unroll 1
    for (int e = 0; e < E_DIM; ++e) {
        v8f c = {};
#pragma unroll
        for (int f0 = 0; f0 < F_DIM; f0 += 32) {
            const float* ap = Abase + (size_t)e * F_DIM + f0;
            const float* bp = Bbase + (size_t)e * F_DIM + f0;
            v16h a = cvt16(*(const float4*)(ap +  0), *(const float4*)(ap +  4),
                           *(const float4*)(ap + 16), *(const float4*)(ap + 20));
            v16h b = cvt16(*(const float4*)(bp +  0), *(const float4*)(bp +  4),
                           *(const float4*)(bp + 16), *(const float4*)(bp + 20));
            c = __builtin_amdgcn_wmma_f32_16x16x32_f16(false, a, false, b,
                                                       (short)0, c, false, false);
        }
        const float muc = mu_stc * (float)e;         // mu_e pre-scaled by cfac
#pragma unroll
        for (int i = 0; i < 8; ++i) {
            const float atn = c[i];
#pragma unroll
            for (int p = 0; p < 8; ++p) {
                const float z = dsc[i][p] - muc;
                acc[p] += atn * __builtin_amdgcn_exp2f(-z * z);
            }
        }
    }

    // ---- wave reduce (all 32 lanes) ----
#pragma unroll
    for (int m = 1; m < 32; m <<= 1) {
#pragma unroll
        for (int p = 0; p < 8; ++p) acc[p] += __shfl_xor(acc[p], m, 32);
    }

    __shared__ float smem[8 * 8];
    if (lane < 8) smem[wave * 8 + lane] = acc[lane];
    __syncthreads();
    if (threadIdx.x < 8) {
        float s = 0.0f;
#pragma unroll
        for (int w = 0; w < 8; ++w) s += smem[w * 8 + threadIdx.x];
        partial[blockIdx.x * 8 + threadIdx.x] = s;
    }
}

__global__ __launch_bounds__(256) void ff_reduce(const float* __restrict__ partial,
                                                 const float* __restrict__ weight,
                                                 const float* __restrict__ bias,
                                                 float* __restrict__ out,
                                                 int nblocks) {
    const int t = threadIdx.x;
    const int p = t & 7;
    const int g = t >> 3;                            // 0..31
    float s = 0.0f;
    for (int j = g; j < nblocks; j += 32) s += partial[j * 8 + p];
    // lanes l, l^8, l^16, l^24 share the same p within a wave
    s += __shfl_xor(s, 8, 32);
    s += __shfl_xor(s, 16, 32);
    __shared__ float sm[64];
    const int wave = t >> 5, lane = t & 31;
    if (lane < 8) sm[wave * 8 + lane] = s;
    __syncthreads();
    if (t < 8) {
        float tot = 0.0f;
#pragma unroll
        for (int w = 0; w < 8; ++w) tot += sm[w * 8 + t];
        out[t] = tot * weight[0] + bias[0];
    }
}

extern "C" void kernel_launch(void* const* d_in, const int* in_sizes, int n_in,
                              void* d_out, int out_size, void* d_ws, size_t ws_size,
                              hipStream_t stream) {
    const float* lig_feat   = (const float*)d_in[0];
    const float* rec_feat   = (const float*)d_in[1];
    const float* lig_coords = (const float*)d_in[2];
    const float* rec_coord  = (const float*)d_in[3];
    const float* weight     = (const float*)d_in[4];
    const float* bias       = (const float*)d_in[5];
    float* out     = (float*)d_out;
    float* partial = (float*)d_ws;                    // 128 * 8 floats = 4 KB

    ff_main<<<128, 256, 0, stream>>>(lig_feat, rec_feat, lig_coords, rec_coord, partial);
    ff_reduce<<<1, 256, 0, stream>>>(partial, weight, bias, out, 128);
}
